// LinearLayer_45243185496808
// MI455X (gfx1250) — matrix operand order repaired and checked by reading, not compiled, not run on hardware
//
#include <hip/hip_runtime.h>
#include <math.h>

typedef __attribute__((ext_vector_type(16))) _Float16 v16h;
typedef __attribute__((ext_vector_type(8)))  float    v8f;
typedef __attribute__((ext_vector_type(4)))  float    v4f;

#define BATCH   131072
#define IN_SZ   128
#define OUT_SZ  128
#define P_NUM   16

__device__ __forceinline__ float fast_tanh(float a) {
#if __has_builtin(__builtin_amdgcn_tanhf)
    return __builtin_amdgcn_tanhf(a);   // v_tanh_f32 (gfx1250 trans op)
#else
    return tanhf(a);
#endif
}

// ---------------------------------------------------------------------------
// Kernel 1: w[o,i] = (f16) sum_p coef[o,i,p]   (128*128 outputs, 16 f32 each)
// ---------------------------------------------------------------------------
__global__ __launch_bounds__(256) void reduce_coef_kernel(
    const float* __restrict__ coef, _Float16* __restrict__ w)
{
    int idx = blockIdx.x * blockDim.x + threadIdx.x;   // 0 .. 16383  (= o*128 + i)
    const float* p = coef + (size_t)idx * P_NUM;
    v4f a = *(const v4f*)(p +  0);
    v4f b = *(const v4f*)(p +  4);
    v4f c = *(const v4f*)(p +  8);
    v4f d = *(const v4f*)(p + 12);
    float s = 0.f;
#pragma unroll
    for (int j = 0; j < 4; ++j) s += a[j] + b[j] + c[j] + d[j];
    w[idx] = (_Float16)s;
}

// ---------------------------------------------------------------------------
// Kernel 2: out = tanh(x*t) @ w^T  via v_wmma_f32_16x16x32_f16
//   block = 256 thr = 8 waves; each wave: 16 rows x 128 cols of output
//   grid  = 131072/128 = 1024 blocks
// ---------------------------------------------------------------------------
__global__ __launch_bounds__(256) void tanh_gemm_kernel(
    const float*    __restrict__ x,
    const _Float16* __restrict__ w,     // [O=128][I=128] f16, row-major
    const float*    __restrict__ trange,
    float*          __restrict__ out)
{
    const int lane    = threadIdx.x & 31;
    const int wave    = threadIdx.x >> 5;
    const int row0    = blockIdx.x * 128 + wave * 16;   // first of this wave's 16 rows
    const int m       = lane & 15;
    const int halfsel = lane >> 4;                      // 0: lanes 0-15, 1: lanes 16-31
    const float t     = trange[0];

    // ---- Build A fragments (16x32 f16, contiguous-k layout) for all 4 K-blocks -----
    // lane<16 : h[0..15] -> K = kb*32 + 0..15
    // lane>=16: h[0..15] -> K = kb*32 + 16..31
    const int kbase = halfsel * 16;
    const float* xrow = x + (size_t)(row0 + m) * IN_SZ;

    v16h A[4];
#pragma unroll
    for (int kb = 0; kb < 4; ++kb) {
        const int k0 = kb * 32 + kbase;
        v4f f0 = *(const v4f*)(xrow + k0 +  0);
        v4f f1 = *(const v4f*)(xrow + k0 +  4);
        v4f f2 = *(const v4f*)(xrow + k0 +  8);
        v4f f3 = *(const v4f*)(xrow + k0 + 12);
        v16h a;
#pragma unroll
        for (int j = 0; j < 4; ++j) {
            a[j +  0] = (_Float16)fast_tanh(f0[j] * t);
            a[j +  4] = (_Float16)fast_tanh(f1[j] * t);
            a[j +  8] = (_Float16)fast_tanh(f2[j] * t);
            a[j + 12] = (_Float16)fast_tanh(f3[j] * t);
        }
        A[kb] = a;
    }

    // ---- 8 N-tiles x 4 K-steps of WMMA, A kept resident ---------------------
    // B fragment (32x16 f16): lane n (n=lane&15) = output column nt*16+n,
    // lanes 0-15 hold K = kb*32 + 0..15, lanes 16-31 hold K = kb*32 + 16..31
    // -> 16 contiguous halfs from w[col][...] (32-byte aligned).
    const int koff = halfsel * 16;

    v8f acc[8];
#pragma unroll
    for (int nt = 0; nt < 8; ++nt) acc[nt] = (v8f){0.f,0.f,0.f,0.f,0.f,0.f,0.f,0.f};

#pragma unroll
    for (int nt = 0; nt < 8; ++nt) {
        const _Float16* wb = w + (size_t)(nt * 16 + m) * IN_SZ + koff;
#pragma unroll
        for (int kb = 0; kb < 4; ++kb) {
            v16h b = *(const v16h*)(wb + kb * 32);
            acc[nt] = __builtin_amdgcn_wmma_f32_16x16x32_f16(
                /*neg_a=*/false, A[kb], /*neg_b=*/false, b,
                /*c_mod=*/(short)0, acc[nt],
                /*reuse_a=*/false, /*reuse_b=*/false);
        }
    }

    // ---- Store D tiles (C/D layout: N = lane&15, M = halfsel*8 + r) --------
#pragma unroll
    for (int nt = 0; nt < 8; ++nt) {
        float* obase = out + (size_t)(row0 + halfsel * 8) * OUT_SZ + nt * 16 + m;
#pragma unroll
        for (int r = 0; r < 8; ++r) {
            obase[(size_t)r * OUT_SZ] = acc[nt][r];
        }
    }
}

// ---------------------------------------------------------------------------
extern "C" void kernel_launch(void* const* d_in, const int* in_sizes, int n_in,
                              void* d_out, int out_size, void* d_ws, size_t ws_size,
                              hipStream_t stream)
{
    const float* x    = (const float*)d_in[0];   // [131072, 128] f32
    const float* coef = (const float*)d_in[1];   // [128, 128, 16] f32
    const float* tr   = (const float*)d_in[2];   // scalar f32
    float*       out  = (float*)d_out;           // [131072, 128] f32
    _Float16*    w    = (_Float16*)d_ws;         // 128*128 f16 = 32 KB scratch

    // 1) reduce coef over p -> f16 weight matrix
    reduce_coef_kernel<<<(OUT_SZ * IN_SZ) / 256, 256, 0, stream>>>(coef, w);

    // 2) fused tanh + WMMA GEMM
    tanh_gemm_kernel<<<BATCH / 128, 256, 0, stream>>>(x, w, tr, out);
}
